// AxialAttention2D_84258668413218
// MI455X (gfx1250) — matrix operand order repaired and checked by reading, not compiled, not run on hardware
//
#include <hip/hip_runtime.h>
#include <hip/hip_bf16.h>
#include <math.h>
#include <stdint.h>

typedef __attribute__((ext_vector_type(16))) _Float16 v16h;
typedef __attribute__((ext_vector_type(8)))  _Float16 v8h;
typedef __attribute__((ext_vector_type(4)))  _Float16 v4h;
typedef __attribute__((ext_vector_type(8)))  float    v8f;
typedef __attribute__((ext_vector_type(4)))  unsigned int v4u;
typedef __attribute__((ext_vector_type(8)))  int      v8i;
typedef __attribute__((ext_vector_type(4)))  int      v4i;

#if defined(__has_builtin)
#  if __has_builtin(__builtin_amdgcn_tensor_load_to_lds) && __has_builtin(__builtin_amdgcn_s_wait_tensorcnt)
#    define HAVE_TDM 1
#  endif
#endif
#ifndef HAVE_TDM
#  define HAVE_TDM 0
#endif

#define CN 128      // channels
#define LN 128      // sequence length of each axial pass
#define NHEADS 4
#define HD 32
#define CHW (CN * 16384)   // C*H*W elements per batch image

// ---- LDS byte offsets (single dynamic-LDS arena, regions aliased by phase) ----
#define OFF_WQ   0         // phase1: wqkv f16 [384][128] = 98304 B
#define OFF_ATT  65536     // phase2 alias: LOG f32 [128][128] at 0, ATT f16 [128][128] at 65536
#define OFF_XT   98304     // phase1: x^T f16 [l][c] = 32768 B ; phase2 alias: OUT_T f16 [l][c]
#define OFF_QT   131072    // q^T (scaled) f16 [l][c]
#define OFF_KT   163840    // k^T f16 [l][c]
#define OFF_V    196608    // v f16 [c][l]
#define OFF_WP   229376    // wproj f16 [o][c]
#define OFF_BIAS 262144    // bias f32 [128]
#define SMEM_BYTES 262656  // 256.5 KB  (<= 320 KB WGP LDS)

// f16 weight blob layout inside d_ws (byte offsets)
#define WS_WQH 0
#define WS_WPH 98304
#define WS_WQW 131072
#define WS_WPW 229376
#define WS_TOTAL_F16 131072   // total f16 elements (262144 B)

// A-fragment (16x32 f16): lane holds row M=lane&15; K = k0+{0..7} in regs 0-3, k0+{16..23} in regs 4-7
__device__ inline v16h lds_load_a(const _Float16* p) {
  v8h lo = *(const v8h*)(p);
  v8h hi = *(const v8h*)(p + 16);
  v16h r;
#pragma unroll
  for (int i = 0; i < 8; ++i) { r[i] = lo[i]; r[i + 8] = hi[i]; }
  return r;
}
// B-fragment (32x16 f16): lane holds col N=lane&15; K = k0+{0..7} in regs 0-3, k0+{16..23} in regs 4-7
__device__ inline v16h lds_load_b(const _Float16* p) {
  v8h lo = *(const v8h*)(p);
  v8h hi = *(const v8h*)(p + 16);
  v16h r;
#pragma unroll
  for (int i = 0; i < 8; ++i) { r[i] = lo[i]; r[i + 8] = hi[i]; }
  return r;
}
__device__ inline v8f wmma16(v16h a, v16h b, v8f c) {
  return __builtin_amdgcn_wmma_f32_16x16x32_f16(false, a, false, b, (short)0, c, false, false);
}

#if HAVE_TDM
// TDM linear copy: global f16 blob -> LDS. Degenerate 2D tensor, data_size=8B,
// tile = [nbytes/8, 1]. D# per CDNA5 ISA ch.8 (group0 128b, group1 256b, groups2/3 zero).
// This toolchain's builtin takes 6 args: (g0 u32x4, g1 i32x8, g2 i32x4, g3 i32x4, i32x8, cpol).
__device__ inline void tdm_load_lin(unsigned lds_off, const void* gptr, unsigned nbytes) {
  unsigned long long ga = (unsigned long long)(uintptr_t)gptr;
  unsigned u = nbytes >> 3;                       // 8-byte units
  v4u g0;
  g0[0] = 1u;                                    // count=1, user mode
  g0[1] = lds_off;                               // lds_addr (bits 63:32)
  g0[2] = (unsigned)(ga & 0xFFFFFFFFull);        // global_addr[31:0]
  g0[3] = (unsigned)((ga >> 32) & 0x1FFFFFFull)  // global_addr[56:32]
        | (2u << 30);                            // type=2 ("image")
  v8i g1;
  g1[0] = (int)(3u << 16);                       // data_size=3 (8B); wg_mask=0
  g1[1] = (int)((u & 0xFFFFu) << 16);            // tensor_dim0[15:0]   @bits63:48
  g1[2] = (int)((u >> 16) | (1u << 16));         // tensor_dim0[31:16], tensor_dim1=1
  g1[3] = (int)((u & 0xFFFFu) << 16);            // tile_dim0           @bits127:112
  g1[4] = 1;                                     // tile_dim1=1, tile_dim2=0
  g1[5] = (int)u;                                // tensor_dim0_stride[31:0]
  g1[6] = (int)((u & 0xFFFFu) << 16);            // stride0[47:32]=0, stride1[15:0]
  g1[7] = (int)(u >> 16);                        // stride1[47:16]
  v4i z4 = {0, 0, 0, 0};
  v8i z8 = {0, 0, 0, 0, 0, 0, 0, 0};
  __builtin_amdgcn_tensor_load_to_lds(g0, g1, z4, z4, z8, 0);
}
#endif

// ---------------- prep: one-time fp32 -> f16 weight conversion into d_ws ----------------
__global__ __launch_bounds__(256)
void convert_weights_kernel(const float* __restrict__ wq_h, const float* __restrict__ wp_h,
                            const float* __restrict__ wq_w, const float* __restrict__ wp_w,
                            _Float16* __restrict__ ws) {
  int e = (blockIdx.x * 256 + threadIdx.x) * 4;   // f16 element index into ws
  if (e >= WS_TOTAL_F16) return;
  const float* src; int off;
  if      (e < 49152)  { src = wq_h; off = 0; }
  else if (e < 65536)  { src = wp_h; off = 49152; }
  else if (e < 114688) { src = wq_w; off = 65536; }
  else                 { src = wp_w; off = 114688; }
  float4 f = *(const float4*)(src + (e - off));
  v4h h4 = { (_Float16)f.x, (_Float16)f.y, (_Float16)f.z, (_Float16)f.w };
  *(v4h*)(ws + e) = h4;
}

__global__ __launch_bounds__(256, 1)
void axial_pass_kernel(const float* __restrict__ x,
                       const _Float16* __restrict__ wq16,  // [384][128] f16
                       const _Float16* __restrict__ wp16,  // [128][128] f16
                       const float* __restrict__ bias,
                       float* __restrict__ out,
                       int lStride,      // 1 = height pass (attend along W), 128 = width pass
                       int accumulate)   // 0 = write 0.5*y, 1 = out += 0.5*y
{
  extern __shared__ char smem[];
  _Float16* WQ    = (_Float16*)(smem + OFF_WQ);
  float*    LOG   = (float*)   (smem + OFF_WQ);    // alias (phase 2)
  _Float16* ATT   = (_Float16*)(smem + OFF_ATT);   // alias (phase 2)
  _Float16* XT    = (_Float16*)(smem + OFF_XT);
  _Float16* OUT_T = (_Float16*)(smem + OFF_XT);    // alias (phase 2)
  _Float16* QT    = (_Float16*)(smem + OFF_QT);
  _Float16* KT    = (_Float16*)(smem + OFF_KT);
  _Float16* Vs    = (_Float16*)(smem + OFF_V);
  _Float16* WP    = (_Float16*)(smem + OFF_WP);
  float*    BIA   = (float*)   (smem + OFF_BIAS);

  const int tid   = threadIdx.x;
  const int lane  = tid & 31;          // wave32
  const int wave  = tid >> 5;          // 8 waves
  const int ln16  = lane & 15;
  const int hi    = (lane >> 4) & 1;
  const int hi8   = hi * 8;            // A-fragment K sub-offset; also D-tile row half
  const int klo16 = hi * 8;            // B-fragment K sub-offset

  // slab: blockIdx = b*128 + p ; p = h (pass0) or w (pass1)
  const int b    = (int)blockIdx.x >> 7;
  const int p    = (int)blockIdx.x & 127;
  const int base = b * CHW + ((lStride == 1) ? p * 128 : p);

  // ---------------- Phase 0: staging ----------------
  // weights: DMA'd by the Tensor Data Mover (f16 blobs, pure byte copy),
  // overlapped with VALU staging of the x slab below.
#if HAVE_TDM
  if (wave == 0) {
    tdm_load_lin(OFF_WQ, wq16, 3 * CN * CN * 2);   // 98304 B
    tdm_load_lin(OFF_WP, wp16, CN * CN * 2);       // 32768 B
  }
#else
  for (int i = tid; i < (3 * CN * CN) / 4; i += 256)
    ((v4h*)(smem + OFF_WQ))[i] = ((const v4h*)wq16)[i];
  for (int i = tid; i < (CN * CN) / 4; i += 256)
    ((v4h*)(smem + OFF_WP))[i] = ((const v4h*)wp16)[i];
#endif
  if (tid < CN) BIA[tid] = bias[tid];
  for (int e = tid; e < CN * LN; e += 256) {
    int c = e >> 7, l = e & 127;
    float v = x[base + c * 16384 + l * lStride];
    XT[l * CN + c] = (_Float16)v;               // x^T for B-fragment friendliness
  }
#if HAVE_TDM
  __builtin_amdgcn_s_wait_tensorcnt(0);
#endif
  __syncthreads();

  // ---------------- Phase 1 (M1): qkv = wqkv(f16) @ x ----------------
  // each wave owns one N(=l) tile; B fragments loaded once, reused over 24 M tiles
  {
    const int nt = wave;
    const _Float16* brow = XT + (nt * 16 + ln16) * CN;
    v16h b0 = lds_load_b(brow + 0 * 32 + klo16);
    v16h b1 = lds_load_b(brow + 1 * 32 + klo16);
    v16h b2 = lds_load_b(brow + 2 * 32 + klo16);
    v16h b3 = lds_load_b(brow + 3 * 32 + klo16);
    const int nIdx = nt * 16 + ln16;
    for (int mt = 0; mt < 24; ++mt) {
      const _Float16* arow = WQ + (mt * 16 + ln16) * CN;
      v8f acc = {};
      acc = wmma16(lds_load_a(arow + 0 * 32 + hi8), b0, acc);
      acc = wmma16(lds_load_a(arow + 1 * 32 + hi8), b1, acc);
      acc = wmma16(lds_load_a(arow + 2 * 32 + hi8), b2, acc);
      acc = wmma16(lds_load_a(arow + 3 * 32 + hi8), b3, acc);
      const int mb = mt * 16 + hi8;
#pragma unroll
      for (int r = 0; r < 8; ++r) {
        int o = mb + r;
        float v = acc[r];
        if (o < CN)            QT[nIdx * CN + o]            = (_Float16)(v * 0.17677669529663688f); // fold hd^-0.5
        else if (o < 2 * CN)   KT[nIdx * CN + (o - CN)]     = (_Float16)v;
        else                   Vs[(o - 2 * CN) * LN + nIdx] = (_Float16)v;
      }
    }
  }
  __syncthreads();

  // ---------------- Phase 2: per-head attention ----------------
  for (int h = 0; h < NHEADS; ++h) {
    // M2: logits[i][j] = sum_d qT[i][d] * k[d][j]  (single K=32 WMMA per tile)
    {
      const int nt = wave;
      v16h bk = lds_load_b(KT + (nt * 16 + ln16) * CN + h * HD + klo16);
      for (int mt = 0; mt < 8; ++mt) {
        v16h aq = lds_load_a(QT + (mt * 16 + ln16) * CN + h * HD + hi8);
        v8f acc = {};
        acc = wmma16(aq, bk, acc);
        const int mb = mt * 16 + hi8;
#pragma unroll
        for (int r = 0; r < 8; ++r)
          LOG[(mb + r) * LN + nt * 16 + ln16] = acc[r];
      }
    }
    __syncthreads();

    // row softmax (128 rows, one thread each)
    if (tid < LN) {
      float* row = LOG + tid * LN;
      float m = row[0];
      for (int j = 1; j < LN; ++j) m = fmaxf(m, row[j]);
      float s = 0.f;
      for (int j = 0; j < LN; ++j) {
        float e = __expf(row[j] - m);
        s += e;
        ATT[tid * LN + j] = (_Float16)e;
      }
      float rinv = 1.0f / s;
      for (int j = 0; j < LN; ++j)
        ATT[tid * LN + j] = (_Float16)((float)ATT[tid * LN + j] * rinv);
    }
    __syncthreads();

    // M3: outT[i][d] = sum_j attn[i][j] * v[d][j]  (K=128 -> 4 WMMAs)
    {
      const int nt = wave & 1;
      const _Float16* vrow = Vs + (h * HD + nt * 16 + ln16) * LN;
      v16h vb0 = lds_load_b(vrow + 0 * 32 + klo16);
      v16h vb1 = lds_load_b(vrow + 1 * 32 + klo16);
      v16h vb2 = lds_load_b(vrow + 2 * 32 + klo16);
      v16h vb3 = lds_load_b(vrow + 3 * 32 + klo16);
      for (int tt = 0; tt < 2; ++tt) {
        const int mt = (wave >> 1) + tt * 4;
        const _Float16* arow = ATT + (mt * 16 + ln16) * LN;
        v8f acc = {};
        acc = wmma16(lds_load_a(arow + 0 * 32 + hi8), vb0, acc);
        acc = wmma16(lds_load_a(arow + 1 * 32 + hi8), vb1, acc);
        acc = wmma16(lds_load_a(arow + 2 * 32 + hi8), vb2, acc);
        acc = wmma16(lds_load_a(arow + 3 * 32 + hi8), vb3, acc);
        const int mb = mt * 16 + hi8;
#pragma unroll
        for (int r = 0; r < 8; ++r)
          OUT_T[(mb + r) * CN + h * HD + nt * 16 + ln16] = (_Float16)acc[r];
      }
    }
    // no barrier here: next head's M2 (writes LOG only) is safe, and its barrier
    // orders all M3 ATT-reads before the next softmax overwrites ATT.
  }
  __syncthreads();

  // ---------------- Phase 3 (M4): y^T = outT @ wproj^T, fused bias + 0.5x + store ----------------
  {
    const int nt = wave;
    const _Float16* wrow = WP + (nt * 16 + ln16) * CN;
    v16h wb0 = lds_load_b(wrow + 0 * 32 + klo16);
    v16h wb1 = lds_load_b(wrow + 1 * 32 + klo16);
    v16h wb2 = lds_load_b(wrow + 2 * 32 + klo16);
    v16h wb3 = lds_load_b(wrow + 3 * 32 + klo16);
    const int o = nt * 16 + ln16;
    const float bia = BIA[o];
    for (int mt = 0; mt < 8; ++mt) {
      const _Float16* arow = OUT_T + (mt * 16 + ln16) * CN;
      v8f acc = {};
      acc = wmma16(lds_load_a(arow + 0 * 32 + hi8), wb0, acc);
      acc = wmma16(lds_load_a(arow + 1 * 32 + hi8), wb1, acc);
      acc = wmma16(lds_load_a(arow + 2 * 32 + hi8), wb2, acc);
      acc = wmma16(lds_load_a(arow + 3 * 32 + hi8), wb3, acc);
      const int mb = mt * 16 + hi8;
      if (!accumulate && lStride == 1) {
        // pass 0: 8 consecutive l -> two b128 stores
        const int idx = base + o * 16384 + mb;
        float4 lo4 = { 0.5f * (acc[0] + bia), 0.5f * (acc[1] + bia),
                       0.5f * (acc[2] + bia), 0.5f * (acc[3] + bia) };
        float4 hi4 = { 0.5f * (acc[4] + bia), 0.5f * (acc[5] + bia),
                       0.5f * (acc[6] + bia), 0.5f * (acc[7] + bia) };
        *(float4*)(out + idx)     = lo4;
        *(float4*)(out + idx + 4) = hi4;
      } else {
#pragma unroll
        for (int r = 0; r < 8; ++r) {
          const int l = mb + r;
          const int idx = base + o * 16384 + l * lStride;
          const float val = 0.5f * (acc[r] + bia);
          if (accumulate) out[idx] = out[idx] + val;
          else            out[idx] = val;
        }
      }
    }
  }
}

extern "C" void kernel_launch(void* const* d_in, const int* in_sizes, int n_in,
                              void* d_out, int out_size, void* d_ws, size_t ws_size,
                              hipStream_t stream) {
  (void)in_sizes; (void)n_in; (void)out_size; (void)ws_size;
  const float* x       = (const float*)d_in[0];
  const float* wqkv_h  = (const float*)d_in[1];
  const float* wproj_h = (const float*)d_in[2];
  const float* bproj_h = (const float*)d_in[3];
  const float* wqkv_w  = (const float*)d_in[4];
  const float* wproj_w = (const float*)d_in[5];
  const float* bproj_w = (const float*)d_in[6];
  float* out = (float*)d_out;
  _Float16* ws = (_Float16*)d_ws;

  // opt in to >64KB dynamic LDS (deterministic, no guards)
  (void)hipFuncSetAttribute((const void*)axial_pass_kernel,
                            hipFuncAttributeMaxDynamicSharedMemorySize, SMEM_BYTES);

  // one-time f32 -> f16 weight conversion into scratch (L2-resident, 256 KB)
  convert_weights_kernel<<<dim3(WS_TOTAL_F16 / 4 / 256), dim3(256), 0, stream>>>(
      wqkv_h, wproj_h, wqkv_w, wproj_w, ws);

  dim3 grid(1024), block(256);
  // height pass: slabs (b,h), attend along W (lStride=1), write 0.5*y
  axial_pass_kernel<<<grid, block, SMEM_BYTES, stream>>>(
      x, (const _Float16*)((const char*)d_ws + WS_WQH),
         (const _Float16*)((const char*)d_ws + WS_WPH), bproj_h, out,
      /*lStride=*/1, /*accumulate=*/0);
  // width pass: slabs (b,w), attend along H (lStride=W=128), accumulate 0.5*y
  axial_pass_kernel<<<grid, block, SMEM_BYTES, stream>>>(
      x, (const _Float16*)((const char*)d_ws + WS_WQW),
         (const _Float16*)((const char*)d_ws + WS_WPW), bproj_w, out,
      /*lStride=*/128, /*accumulate=*/1);
}
